// PCAGATLayer_644245094860
// MI455X (gfx1250) — matrix operand order repaired and checked by reading, not compiled, not run on hardware
//
#include <hip/hip_runtime.h>
#include <hip/hip_bf16.h>
#include <float.h>

// ---------- types ----------
typedef __attribute__((ext_vector_type(16))) _Float16 v16h;
typedef __attribute__((ext_vector_type(8)))  _Float16 v8h;
typedef __attribute__((ext_vector_type(8)))  float    v8f;

#define D     128
#define D3    384
#define D2    256
#define AW1_STRIDE (D3 + 8)   // padded LDS row stride: 196 DWORDs == 4 mod 64 banks
#define HBITS 17
#define HSZ   (1u << HBITS)
#define HMASK (HSZ - 1u)
#define EMPTYKEY 0xFFFFFFFFu

__device__ __forceinline__ float leakyf(float x) { return x >= 0.f ? x : 0.2f * x; }
__device__ __forceinline__ float sigmoidf_(float x) { return 1.f / (1.f + __expf(-x)); }

__device__ __forceinline__ void atomicMaxF(float* addr, float v) {
  if (v >= 0.f) atomicMax((int*)addr, __float_as_int(v));
  else          atomicMin((unsigned int*)addr, __float_as_uint(v));
}

__device__ __forceinline__ v16h pack16(v8h lo, v8h hi) {
  v16h a;
#pragma unroll
  for (int j = 0; j < 8; ++j) { a[j] = lo[j]; a[8 + j] = hi[j]; }
  return a;
}

// ---------- small utility kernels ----------
__global__ void k_cvt(const float* __restrict__ s, _Float16* __restrict__ d, int nElem) {
  int i = blockIdx.x * blockDim.x + threadIdx.x;
  if (i < nElem) d[i] = (_Float16)s[i];
}

__global__ void k_init(float* side, float* mx, float* sm,
                       unsigned* hkeys, float* hagg, float* hcnt,
                       unsigned* scal, int n) {
  int i = blockIdx.x * blockDim.x + threadIdx.x;
  int nd = n * D;
  if (i < nd) side[i] = 0.f;
  if (i < n) { mx[i] = -FLT_MAX; sm[i] = 0.f; }
  if (i < (int)HSZ) { hkeys[i] = EMPTYKEY; hagg[i] = 0.f; hcnt[i] = 0.f; }
  if (i < 4) scal[i] = 0u;
}

// insert constraints into hash table; accumulate sum of weighted values
__global__ void k_insert(const int* __restrict__ ci, const float* __restrict__ cv,
                         const int* __restrict__ ct, const float* __restrict__ lambda_c,
                         unsigned* hkeys, float* hagg, float* hcnt, float* wssum,
                         int C, int n) {
  int i = blockIdx.x * blockDim.x + threadIdx.x;
  if (i >= C) return;
  unsigned key = (unsigned)(ci[i] * n + ci[C + i]);
  float lc = sigmoidf_(lambda_c[ct[i]]) * 0.9f + 0.1f;
  float w = lc * cv[i];
  atomicAdd(wssum, w);
  unsigned slot = (key * 2654435761u) & HMASK;
  for (unsigned p = 0; p < HSZ; ++p) {
    unsigned old = atomicCAS(&hkeys[slot], EMPTYKEY, key);
    if (old == EMPTYKEY || old == key) {
      atomicAdd(&hagg[slot], w);
      atomicAdd(&hcnt[slot], 1.f);
      return;
    }
    slot = (slot + 1u) & HMASK;
  }
}

// per-edge lookup of aggregated constraint value + existence mask
__global__ void k_probe(const int* __restrict__ ei, const unsigned* __restrict__ hkeys,
                        const float* __restrict__ hagg,
                        float* __restrict__ ms, int* __restrict__ maskb,
                        unsigned* nmatched, int E, int n) {
  int i = blockIdx.x * blockDim.x + threadIdx.x;
  if (i >= E) return;
  unsigned key = (unsigned)(ei[i] * n + ei[E + i]);
  unsigned slot = (key * 2654435761u) & HMASK;
  float agg = 0.f; int m = 0;
  for (unsigned p = 0; p < HSZ; ++p) {
    unsigned k2 = hkeys[slot];
    if (k2 == key) { agg = hagg[slot]; m = 1; break; }
    if (k2 == EMPTYKEY) break;
    slot = (slot + 1u) & HMASK;
  }
  ms[i] = m ? agg : 0.f;
  maskb[i] = m;
  if (m) atomicAdd(nmatched, 1u);
}

// ---------- Wh = h @ W.T  (WMMA f16 -> f32), dual-precision output ----------
__global__ __launch_bounds__(128) void k_wh(const _Float16* __restrict__ hh,
                                            const _Float16* __restrict__ Wf,
                                            float* __restrict__ Wh,
                                            _Float16* __restrict__ Whh, int n) {
  int wave = (blockIdx.x * blockDim.x + threadIdx.x) >> 5;
  int lane = threadIdx.x & 31;
  int mtiles = n >> 4;
  int tm = wave >> 3, tn = wave & 7;          // 8 N-tiles of 16 in D=128
  if (tm >= mtiles) return;                    // wave-uniform guard
  int row = lane & 15, hi = lane >> 4;
  int grow = tm * 16 + row;
  const _Float16* arow = hh + (size_t)grow * D;
  const _Float16* brow = Wf + (size_t)(tn * 16 + row) * D;   // B[k][n] = W[n][k]
  v8f c0 = {}, c1 = {};                        // two chains for XDL ILP
#pragma unroll
  for (int kt = 0; kt < 4; kt += 2) {
    int ka0 = kt * 32 + hi * 16;
    int ka1 = (kt + 1) * 32 + hi * 16;
    v16h a0 = pack16(*(const v8h*)(arow + ka0), *(const v8h*)(arow + ka0 + 8));
    v16h a1 = pack16(*(const v8h*)(arow + ka1), *(const v8h*)(arow + ka1 + 8));
    v16h b0 = *(const v16h*)(brow + kt * 32 + hi * 16);
    v16h b1 = *(const v16h*)(brow + (kt + 1) * 32 + hi * 16);
    c0 = __builtin_amdgcn_wmma_f32_16x16x32_f16(false, a0, false, b0, (short)0, c0, false, false);
    c1 = __builtin_amdgcn_wmma_f32_16x16x32_f16(false, a1, false, b1, (short)0, c1, false, false);
  }
  int ncol = tn * 16 + row;
#pragma unroll
  for (int r = 0; r < 8; ++r) {
    int m = tm * 16 + r + 8 * hi;
    float v = c0[r] + c1[r];
    Wh [(size_t)m * D + ncol] = v;
    Whh[(size_t)m * D + ncol] = (_Float16)v;
  }
}

// ---------- fused edge attention + gate MLPs + epilogue ----------
// one wave handles 16 edges; 96 attention WMMAs + 64 gate WMMAs per wave.
// att_w1 (96KB, too big for the 64KB WGP$) is staged once per block into the
// 320KB LDS with a +16B row pad (conflict-free ds reads); gate_w1 (64KB) fits
// WGP$ and stays on the global path.
extern __shared__ _Float16 s_aw1[];

__global__ __launch_bounds__(256) void k_att(
    const int* __restrict__ ei, const int* __restrict__ etype,
    const _Float16* __restrict__ Whh, const _Float16* __restrict__ relf,
    const _Float16* __restrict__ aw1, const float* __restrict__ ab1,
    const float* __restrict__ aw2,
    const _Float16* __restrict__ gw1, const float* __restrict__ gb1,
    const float* __restrict__ gw2, const float* __restrict__ gb2,
    const float* __restrict__ lambda_g,
    const float* __restrict__ ms_in, const int* __restrict__ maskb,
    const unsigned* __restrict__ nmatched, const float* __restrict__ wssum,
    float* __restrict__ e_out, float* mx, int E, int n, int C) {
  __shared__ float e_sh[8 * 16];
  __shared__ float g_sh[8 * 16];

  // cooperative stage: att_w1 (128 x 384 f16) -> padded LDS (stride 392)
  for (int cch = threadIdx.x; cch < D * (D3 / 8); cch += blockDim.x) {
    int r = cch / (D3 / 8), q = cch % (D3 / 8);
    *(int4*)(s_aw1 + (size_t)r * AW1_STRIDE + q * 8) =
        *(const int4*)(aw1 + (size_t)r * D3 + q * 8);
  }
  __syncthreads();

  int lane = threadIdx.x & 31;
  int wv = threadIdx.x >> 5;
  int base = (blockIdx.x * 8 + wv) * 16;
  int row = lane & 15, hi = lane >> 4;
  int eidx = base + row;
  int src = ei[eidx], tgt = ei[E + eidx], rt = etype[eidx];
  const _Float16* seg0 = Whh + (size_t)src * D;
  const _Float16* seg1 = Whh + (size_t)tgt * D;
  const _Float16* seg2 = relf + (size_t)rt * D;

  // register-resident A: 12 K-tiles of [src(128) | tgt(128) | rel(128)]
  v16h A[12];
#pragma unroll
  for (int kt = 0; kt < 12; ++kt) {
    const _Float16* seg = (kt < 4) ? seg0 : ((kt < 8) ? seg1 : seg2);
    int kk = (kt & 3) * 32 + hi * 16;
    A[kt] = pack16(*(const v8h*)(seg + kk), *(const v8h*)(seg + kk + 8));
  }

  // attention: e[m] = sum_n leaky( (att_in @ att_w1.T)[m][n] + b1[n] ) * att_w2[n]
  float s[8];
#pragma unroll
  for (int r = 0; r < 8; ++r) s[r] = 0.f;
  for (int nt = 0; nt < 8; ++nt) {
    int ncol = nt * 16 + row;
    const _Float16* bb = s_aw1 + (size_t)ncol * AW1_STRIDE + hi * 16;
    v8f c0 = {}, c1 = {};
#pragma unroll
    for (int kt = 0; kt < 12; kt += 2) {
      v16h b0 = pack16(*(const v8h*)(bb + kt * 32), *(const v8h*)(bb + kt * 32 + 8));
      v16h b1 = pack16(*(const v8h*)(bb + (kt + 1) * 32), *(const v8h*)(bb + (kt + 1) * 32 + 8));
      c0 = __builtin_amdgcn_wmma_f32_16x16x32_f16(false, A[kt],     false, b0, (short)0, c0, false, false);
      c1 = __builtin_amdgcn_wmma_f32_16x16x32_f16(false, A[kt + 1], false, b1, (short)0, c1, false, false);
    }
    float b1v = ab1[ncol], w2 = aw2[ncol];
#pragma unroll
    for (int r = 0; r < 8; ++r) s[r] += leakyf(c0[r] + c1[r] + b1v) * w2;
  }
#pragma unroll
  for (int r = 0; r < 8; ++r)
    for (int off = 1; off < 16; off <<= 1) s[r] += __shfl_xor(s[r], off, 32);
  if (row == 0) {
#pragma unroll
    for (int r = 0; r < 8; ++r) e_sh[wv * 16 + hi * 8 + r] = s[r];
  }

  // gate: reuses A[0..7] (gi = [src_emb, tgt_emb]); gate_w1 streamed via WGP$
  float t[8];
#pragma unroll
  for (int r = 0; r < 8; ++r) t[r] = 0.f;
  for (int nt = 0; nt < 8; ++nt) {
    int ncol = nt * 16 + row;
    const _Float16* bb = gw1 + (size_t)ncol * D2 + hi * 16;
    v8f c0 = {}, c1 = {};
#pragma unroll
    for (int kt = 0; kt < 8; kt += 2) {
      v16h b0 = *(const v16h*)(bb + kt * 32);
      v16h b1 = *(const v16h*)(bb + (kt + 1) * 32);
      c0 = __builtin_amdgcn_wmma_f32_16x16x32_f16(false, A[kt],     false, b0, (short)0, c0, false, false);
      c1 = __builtin_amdgcn_wmma_f32_16x16x32_f16(false, A[kt + 1], false, b1, (short)0, c1, false, false);
    }
    float b1v = gb1[ncol], w2 = gw2[ncol];
#pragma unroll
    for (int r = 0; r < 8; ++r) t[r] += leakyf(c0[r] + c1[r] + b1v) * w2;
  }
#pragma unroll
  for (int r = 0; r < 8; ++r)
    for (int off = 1; off < 16; off <<= 1) t[r] += __shfl_xor(t[r], off, 32);
  if (row == 0) {
#pragma unroll
    for (int r = 0; r < 8; ++r) g_sh[wv * 16 + hi * 8 + r] = t[r];
  }
  __syncthreads();

  if (hi == 0) {   // 16 lanes, one edge each
    float e = e_sh[wv * 16 + row];
    float g = sigmoidf_(g_sh[wv * 16 + row] + gb2[0]);
    float lg = sigmoidf_(lambda_g[0]) * (1.0f - 0.1f) + 0.1f;
    float msv = ms_in[eidx];
    float msf = maskb[eidx] ? g * msv : 0.f;
    if (*nmatched > 0u) e += lg * msf;
    else                e += lg * (wssum[0] / (float)C) * 0.01f;
    e_out[eidx] = e;
    atomicMaxF(&mx[src], e);
  }
}

// ---------- softmax pieces ----------
__global__ void k_ex(const float* __restrict__ e, const int* __restrict__ ei,
                     const float* __restrict__ mx, float* __restrict__ ex,
                     float* sm, int E) {
  int i = blockIdx.x * blockDim.x + threadIdx.x;
  if (i >= E) return;
  int s = ei[i];
  float v = __expf(fminf(e[i] - mx[s], 20.f));
  ex[i] = v;
  atomicAdd(&sm[s], v);
}

// side[src] += attn * Wh[tgt]; one thread per (edge, 4 columns)
__global__ void k_side(const float* __restrict__ ex, const float* __restrict__ sm,
                       const int* __restrict__ ei, const float* __restrict__ Wh,
                       float* side, int E) {
  int gid = blockIdx.x * blockDim.x + threadIdx.x;
  int e = gid >> 5;
  if (e >= E) return;
  int cs = (gid & 31) * 4;
  int src = ei[e], tgt = ei[E + e];
  float attn = ex[e] / (sm[src] + 1e-10f);
  const float4 tv = *(const float4*)(Wh + (size_t)tgt * D + cs);
  float* sp = side + (size_t)src * D + cs;
  atomicAdd(sp + 0, attn * tv.x);
  atomicAdd(sp + 1, attn * tv.y);
  atomicAdd(sp + 2, attn * tv.z);
  atomicAdd(sp + 3, attn * tv.w);
}

// ---------- final: elu( leaky(Wh+side) + leaky((Wh*side)@W_bi.T) + Wh ) ----------
__global__ __launch_bounds__(128) void k_final(const float* __restrict__ Wh,
                                               const float* __restrict__ side,
                                               const _Float16* __restrict__ Wbif,
                                               float* __restrict__ out, int n) {
  int wave = (blockIdx.x * blockDim.x + threadIdx.x) >> 5;
  int lane = threadIdx.x & 31;
  int mtiles = n >> 4;
  int tm = wave >> 3, tn = wave & 7;
  if (tm >= mtiles) return;
  int row = lane & 15, hi = lane >> 4;
  int grow = tm * 16 + row;
  const float* ar = Wh + (size_t)grow * D;
  const float* sr = side + (size_t)grow * D;
  const _Float16* br = Wbif + (size_t)(tn * 16 + row) * D;
  v8f c0 = {}, c1 = {};
#pragma unroll
  for (int kt = 0; kt < 4; kt += 2) {
#pragma unroll
    for (int u = 0; u < 2; ++u) {
      int ka = (kt + u) * 32 + hi * 16;
      v16h a;
#pragma unroll
      for (int j = 0; j < 8; ++j) {
        a[j]     = (_Float16)(ar[ka + j]      * sr[ka + j]);
        a[8 + j] = (_Float16)(ar[ka + 8 + j] * sr[ka + 8 + j]);
      }
      v16h b = *(const v16h*)(br + (kt + u) * 32 + hi * 16);
      if (u == 0) c0 = __builtin_amdgcn_wmma_f32_16x16x32_f16(false, a, false, b, (short)0, c0, false, false);
      else        c1 = __builtin_amdgcn_wmma_f32_16x16x32_f16(false, a, false, b, (short)0, c1, false, false);
    }
  }
  int ncol = tn * 16 + row;
#pragma unroll
  for (int r = 0; r < 8; ++r) {
    int m = tm * 16 + r + 8 * hi;
    float wh = Wh[(size_t)m * D + ncol];
    float sd = side[(size_t)m * D + ncol];
    float x = leakyf(wh + sd) + leakyf(c0[r] + c1[r]) + wh;
    out[(size_t)m * D + ncol] = x > 0.f ? x : __expf(x) - 1.f;
  }
}

// ---------- host launcher ----------
extern "C" void kernel_launch(void* const* d_in, const int* in_sizes, int n_in,
                              void* d_out, int out_size, void* d_ws, size_t ws_size,
                              hipStream_t stream) {
  const float* h        = (const float*)d_in[0];
  const int*   ei       = (const int*)d_in[1];
  const int*   etype    = (const int*)d_in[2];
  const float* rel      = (const float*)d_in[3];
  const int*   ci       = (const int*)d_in[4];
  const float* cv       = (const float*)d_in[5];
  const int*   ct       = (const int*)d_in[6];
  const float* W        = (const float*)d_in[7];
  const float* W_bi     = (const float*)d_in[8];
  const float* att_w1   = (const float*)d_in[9];
  const float* att_b1   = (const float*)d_in[10];
  const float* att_w2   = (const float*)d_in[11];
  const float* gate_w1  = (const float*)d_in[12];
  const float* gate_b1  = (const float*)d_in[13];
  const float* gate_w2  = (const float*)d_in[14];
  const float* gate_b2  = (const float*)d_in[15];
  const float* lambda_c = (const float*)d_in[16];
  const float* lambda_g = (const float*)d_in[17];

  const int n = in_sizes[0] / D;
  const int E = in_sizes[2];
  const int C = in_sizes[5];
  const int R = in_sizes[3] / D;
  const size_t nD = (size_t)n * D;

  // workspace bump allocator (256B aligned)
  char* p = (char*)d_ws;
  auto alloc = [&](size_t bytes) -> void* {
    void* r = (void*)p;
    p += (bytes + 255) & ~(size_t)255;
    return r;
  };
  float*     Wh    = (float*)alloc(nD * 4);
  float*     side  = (float*)alloc(nD * 4);
  _Float16*  hh    = (_Float16*)alloc(nD * 2);
  _Float16*  Whh   = (_Float16*)alloc(nD * 2);
  _Float16*  Wf    = (_Float16*)alloc((size_t)D * D * 2);
  _Float16*  Wbif  = (_Float16*)alloc((size_t)D * D * 2);
  _Float16*  aw1f  = (_Float16*)alloc((size_t)D * D3 * 2);
  _Float16*  gw1f  = (_Float16*)alloc((size_t)D * D2 * 2);
  _Float16*  relf  = (_Float16*)alloc((size_t)R * D * 2);
  float*     e_buf = (float*)alloc((size_t)E * 4);
  float*     ex_buf= (float*)alloc((size_t)E * 4);
  float*     ms_buf= (float*)alloc((size_t)E * 4);
  int*       maskb = (int*)alloc((size_t)E * 4);
  float*     mx    = (float*)alloc((size_t)n * 4);
  float*     sm    = (float*)alloc((size_t)n * 4);
  unsigned*  hkeys = (unsigned*)alloc((size_t)HSZ * 4);
  float*     hagg  = (float*)alloc((size_t)HSZ * 4);
  float*     hcnt  = (float*)alloc((size_t)HSZ * 4);
  unsigned*  scal  = (unsigned*)alloc(256);
  unsigned*  nmatched = scal;               // scal[0]
  float*     wssum    = (float*)(scal + 1); // scal[1]

  const int TB = 256;
  // 1. f32 -> f16 conversions
  k_cvt<<<((int)nD + TB - 1) / TB, TB, 0, stream>>>(h, hh, (int)nD);
  k_cvt<<<(D * D + TB - 1) / TB, TB, 0, stream>>>(W, Wf, D * D);
  k_cvt<<<(D * D + TB - 1) / TB, TB, 0, stream>>>(W_bi, Wbif, D * D);
  k_cvt<<<(D * D3 + TB - 1) / TB, TB, 0, stream>>>(att_w1, aw1f, D * D3);
  k_cvt<<<(D * D2 + TB - 1) / TB, TB, 0, stream>>>(gate_w1, gw1f, D * D2);
  k_cvt<<<(R * D + TB - 1) / TB, TB, 0, stream>>>(rel, relf, R * D);

  // 2. init accumulators / hash
  int initN = (int)nD > (int)HSZ ? (int)nD : (int)HSZ;
  k_init<<<(initN + TB - 1) / TB, TB, 0, stream>>>(side, mx, sm, hkeys, hagg, hcnt, scal, n);

  // 3-4. constraint hash insert + per-edge probe
  k_insert<<<(C + TB - 1) / TB, TB, 0, stream>>>(ci, cv, ct, lambda_c,
                                                 hkeys, hagg, hcnt, wssum, C, n);
  k_probe<<<(E + TB - 1) / TB, TB, 0, stream>>>(ei, hkeys, hagg, ms_buf, maskb, nmatched, E, n);

  // 5. Wh GEMM (WMMA)
  int waves = (n / 16) * 8;
  k_wh<<<(waves + 3) / 4, 128, 0, stream>>>(hh, Wf, Wh, Whh, n);

  // 6. fused edge attention + gate (WMMA, att_w1 staged in LDS); E % 128 == 0
  size_t att_lds = (size_t)D * AW1_STRIDE * sizeof(_Float16);  // 100352 B
  k_att<<<E / 128, 256, att_lds, stream>>>(ei, etype, Whh, relf,
                                           aw1f, att_b1, att_w2,
                                           gw1f, gate_b1, gate_w2, gate_b2,
                                           lambda_g, ms_buf, maskb, nmatched, wssum,
                                           e_buf, mx, E, n, C);

  // 7. softmax numerator + denominator
  k_ex<<<(E + TB - 1) / TB, TB, 0, stream>>>(e_buf, ei, mx, ex_buf, sm, E);

  // 8. weighted neighbor aggregation
  k_side<<<(E * 32 + TB - 1) / TB, TB, 0, stream>>>(ex_buf, sm, ei, Wh, side, E);

  // 9. final bilinear WMMA + activation fusion
  k_final<<<(waves + 3) / 4, 128, 0, stream>>>(Wh, side, Wbif, (float*)d_out, n);
}
